// EGNN_15135464751163
// MI455X (gfx1250) — compile-verified, matrix operand order checked by reading, not run on hardware
//
#include <hip/hip_runtime.h>

typedef __attribute__((ext_vector_type(16))) __bf16 v16bf;
typedef __attribute__((ext_vector_type(8)))  __bf16 v8bf;
typedef __attribute__((ext_vector_type(4)))  __bf16 v4bf;
typedef __attribute__((ext_vector_type(8)))  float  v8f;

#define NNODES 16384
#define KNBR 10
#define HDIM 128
#define NLAYERS 7

// ---------------- device helpers ----------------

__device__ __forceinline__ float siluf(float x) { return x / (1.f + __expf(-x)); }
__device__ __forceinline__ float clip1k(float x) { return fminf(1000.f, fmaxf(-1000.f, x)); }

__device__ __forceinline__ v16bf pack16(v8bf c0, v8bf c1) {
  v16bf r;
#pragma unroll
  for (int i = 0; i < 8; ++i) { r[i] = c0[i]; r[i + 8] = c1[i]; }
  return r;
}

// Load one 16x32 bf16 WMMA fragment slice from an LDS row:
// chunk0 = row[kb .. kb+7], chunk1 = row[kb+16 .. kb+23]   (A-layout pattern)
__device__ __forceinline__ v16bf ldsFragRow(const __bf16* row, int kb) {
  v8bf c0 = *(const v8bf*)(row + kb);
  v8bf c1 = *(const v8bf*)(row + kb + 16);
  return pack16(c0, c1);
}

// Build a bf16 fragment from two 8-float global chunks
__device__ __forceinline__ v16bf gblFragF32(const float* p0, const float* p1) {
  float4 a0 = *(const float4*)p0;
  float4 a1 = *(const float4*)(p0 + 4);
  float4 b0 = *(const float4*)p1;
  float4 b1 = *(const float4*)(p1 + 4);
  v16bf r;
  r[0] = (__bf16)a0.x; r[1] = (__bf16)a0.y; r[2] = (__bf16)a0.z; r[3] = (__bf16)a0.w;
  r[4] = (__bf16)a1.x; r[5] = (__bf16)a1.y; r[6] = (__bf16)a1.z; r[7] = (__bf16)a1.w;
  r[8]  = (__bf16)b0.x; r[9]  = (__bf16)b0.y; r[10] = (__bf16)b0.z; r[11] = (__bf16)b0.w;
  r[12] = (__bf16)b1.x; r[13] = (__bf16)b1.y; r[14] = (__bf16)b1.z; r[15] = (__bf16)b1.w;
  return r;
}

__device__ __forceinline__ v8f wmma_bf16(v16bf a, v16bf b, v8f c) {
  return __builtin_amdgcn_wmma_f32_16x16x32_bf16(false, a, false, b, (short)0, c, false, false);
}

#define WAIT_DS() asm volatile("s_wait_dscnt 0" ::: "memory")

// ---------------- LDS layout (edge kernel) ----------------
// 0       : w1t  bf16[128][256]  (edge_w1 rows 1..256, transposed, col-major by n)
// 65536   : w2t  bf16[128][128]
// 98304   : c1t  bf16[128][128]
// 131072  : cw2b bf16[128]
// 131328  : w1r  f32[128]   (edge_w1 row 0 = radial weights)
// 131840  : eb1  f32[128]
// 132352  : eb2  f32[128]
// 132864  : cb1  f32[128]
// 133376  : stg  bf16[8 waves][16][128]
// 166144  : rc   f32 [8 waves][16]
// 166656  : nxa  f32 [8 waves][32]
// 167680  : efl  f32 [8 waves][8][128]
#define EDGE_LDS 200448
#define NODE_LDS 132096

__global__ void __launch_bounds__(256)
egnn_edge_kernel(const float* __restrict__ xA, float* __restrict__ xB,
                 const float* __restrict__ hhA, float* __restrict__ efSum,
                 const int* __restrict__ idx,
                 const float* __restrict__ ew1, const float* __restrict__ eb1g,
                 const float* __restrict__ ew2, const float* __restrict__ eb2g,
                 const float* __restrict__ cw1, const float* __restrict__ cb1g,
                 const float* __restrict__ cw2g) {
  extern __shared__ char smem[];
  __bf16* w1t  = (__bf16*)(smem);
  __bf16* w2t  = (__bf16*)(smem + 65536);
  __bf16* c1t  = (__bf16*)(smem + 98304);
  __bf16* cw2b = (__bf16*)(smem + 131072);
  float*  w1r  = (float*)(smem + 131328);
  float*  eb1  = (float*)(smem + 131840);
  float*  eb2  = (float*)(smem + 132352);
  float*  cb1  = (float*)(smem + 132864);
  __bf16* stgA = (__bf16*)(smem + 133376);
  float*  rcA  = (float*)(smem + 166144);
  float*  nxA  = (float*)(smem + 166656);
  float4* eflA = (float4*)(smem + 167680);

  const int tid = threadIdx.x;
  // cooperative weight load + transpose + bf16 convert
  for (int i = tid; i < 128 * 256; i += 256) {
    int n = i >> 8, k = i & 255;
    w1t[i] = (__bf16)ew1[(k + 1) * 128 + n];
  }
  for (int i = tid; i < 128 * 128; i += 256) {
    int n = i >> 7, k = i & 127;
    w2t[i] = (__bf16)ew2[k * 128 + n];
    c1t[i] = (__bf16)cw1[k * 128 + n];
  }
  if (tid < 128) {
    cw2b[tid] = (__bf16)cw2g[tid];
    w1r[tid]  = ew1[tid];
    eb1[tid]  = eb1g[tid];
    eb2[tid]  = eb2g[tid];
    cb1[tid]  = cb1g[tid];
  }
  __syncthreads();

  const int wv = tid >> 5;
  const int lane = tid & 31;
  const int m = lane & 15;
  const int half = lane >> 4;
  __bf16* stg = stgA + wv * (16 * 128);
  float* rc = rcA + wv * 16;
  float* nxa = nxA + wv * 32;
  float4* efl = eflA + wv * 256;  // [8 nodes][32 float4]

  const int g = blockIdx.x * 8 + wv;  // node-group of 8 nodes, 80 edges

  nxa[lane] = 0.f;
#pragma unroll
  for (int i = 0; i < 8; ++i) efl[i * 32 + lane] = make_float4(0.f, 0.f, 0.f, 0.f);

  for (int t = 0; t < 5; ++t) {
    const int e = g * 80 + t * 16 + m;
    const int dst = e / KNBR;
    const int src = idx[dst * KNBR + (e - dst * KNBR)];
    const float dx = xA[src * 3 + 0] - xA[dst * 3 + 0];
    const float dy = xA[src * 3 + 1] - xA[dst * 3 + 1];
    const float dz = xA[src * 3 + 2] - xA[dst * 3 + 2];
    const float radial = dx * dx + dy * dy + dz * dz;
    if (half == 0) rc[m] = radial;
    WAIT_DS();
    float radv[8];
#pragma unroll
    for (int v = 0; v < 8; ++v) radv[v] = rc[v + 8 * half];

    // ---- GEMM1: ef1 = silu([radial|hh_s|hh_d] @ edge_w1 + b1) ----
    v8f acc[8];
#pragma unroll
    for (int nt = 0; nt < 8; ++nt) {
      const int n = nt * 16 + m;
      const float b0 = eb1[n], wr = w1r[n];
#pragma unroll
      for (int v = 0; v < 8; ++v) acc[nt][v] = b0 + radv[v] * wr;  // radial rank-1 term
    }
#pragma unroll
    for (int s = 0; s < 8; ++s) {
      const int kb = s * 32 + half * 8;
      const int j0 = kb, j1 = kb + 16;
      const float* p0 = (j0 < 128) ? (hhA + (size_t)src * 128 + j0)
                                   : (hhA + (size_t)dst * 128 + (j0 - 128));
      const float* p1 = (j1 < 128) ? (hhA + (size_t)src * 128 + j1)
                                   : (hhA + (size_t)dst * 128 + (j1 - 128));
      v16bf a = gblFragF32(p0, p1);
      v16bf bfr[8];
#pragma unroll
      for (int nt = 0; nt < 8; ++nt) bfr[nt] = ldsFragRow(w1t + (nt * 16 + m) * 256, kb);
#pragma unroll
      for (int nt = 0; nt < 8; ++nt) acc[nt] = wmma_bf16(a, bfr[nt], acc[nt]);
    }
#pragma unroll
    for (int nt = 0; nt < 8; ++nt) {
      const int n = nt * 16 + m;
#pragma unroll
      for (int v = 0; v < 8; ++v) stg[(v + 8 * half) * 128 + n] = (__bf16)siluf(acc[nt][v]);
    }
    WAIT_DS();

    // ---- GEMM2: ef = silu(ef1 @ edge_w2 + b2) ----
#pragma unroll
    for (int nt = 0; nt < 8; ++nt) {
      const float b0 = eb2[nt * 16 + m];
#pragma unroll
      for (int v = 0; v < 8; ++v) acc[nt][v] = b0;
    }
#pragma unroll
    for (int s = 0; s < 4; ++s) {
      const int kb = s * 32 + half * 8;
      v16bf a = ldsFragRow(stg + m * 128, kb);
      v16bf bfr[8];
#pragma unroll
      for (int nt = 0; nt < 8; ++nt) bfr[nt] = ldsFragRow(w2t + (nt * 16 + m) * 128, kb);
#pragma unroll
      for (int nt = 0; nt < 8; ++nt) acc[nt] = wmma_bf16(a, bfr[nt], acc[nt]);
    }
    WAIT_DS();
#pragma unroll
    for (int nt = 0; nt < 8; ++nt) {
      const int n = nt * 16 + m;
#pragma unroll
      for (int v = 0; v < 8; ++v) stg[(v + 8 * half) * 128 + n] = (__bf16)siluf(acc[nt][v]);
    }
    WAIT_DS();

    // ef_sum accumulate into per-wave LDS (lane owns columns lane*4..lane*4+3)
#pragma unroll
    for (int r = 0; r < 16; ++r) {
      const int node = (t * 16 + r) / KNBR;
      v4bf ev = *(const v4bf*)(stg + r * 128 + lane * 4);
      float4 s4 = efl[node * 32 + lane];
      s4.x += (float)ev[0]; s4.y += (float)ev[1]; s4.z += (float)ev[2]; s4.w += (float)ev[3];
      efl[node * 32 + lane] = s4;
    }

    // ---- GEMM3: cm1 = silu(ef @ coord_w1 + cb1) ----
#pragma unroll
    for (int nt = 0; nt < 8; ++nt) {
      const float b0 = cb1[nt * 16 + m];
#pragma unroll
      for (int v = 0; v < 8; ++v) acc[nt][v] = b0;
    }
#pragma unroll
    for (int s = 0; s < 4; ++s) {
      const int kb = s * 32 + half * 8;
      v16bf a = ldsFragRow(stg + m * 128, kb);
      v16bf bfr[8];
#pragma unroll
      for (int nt = 0; nt < 8; ++nt) bfr[nt] = ldsFragRow(c1t + (nt * 16 + m) * 128, kb);
#pragma unroll
      for (int nt = 0; nt < 8; ++nt) acc[nt] = wmma_bf16(a, bfr[nt], acc[nt]);
    }
    WAIT_DS();
#pragma unroll
    for (int nt = 0; nt < 8; ++nt) {
      const int n = nt * 16 + m;
#pragma unroll
      for (int v = 0; v < 8; ++v) stg[(v + 8 * half) * 128 + n] = (__bf16)siluf(acc[nt][v]);
    }
    WAIT_DS();

    // ---- GEMV via WMMA: cm = cm1 @ coord_w2 (B columns broadcast) ----
    v8f cacc = {0.f, 0.f, 0.f, 0.f, 0.f, 0.f, 0.f, 0.f};
#pragma unroll
    for (int s = 0; s < 4; ++s) {
      const int kb = s * 32 + half * 8;
      v16bf a = ldsFragRow(stg + m * 128, kb);
      v16bf b = ldsFragRow(cw2b, kb);  // same column vector in all 16 N slots
      cacc = wmma_bf16(a, b, cacc);
    }
    if (m == 0) {
#pragma unroll
      for (int v = 0; v < 8; ++v) rc[v + 8 * half] = cacc[v];
    }
    WAIT_DS();
    if (lane < 16) {
      const float cm = rc[lane];
      const int node = (t * 16 + lane) / KNBR;
      atomicAdd(&nxa[node * 4 + 0], clip1k(dx * cm));
      atomicAdd(&nxa[node * 4 + 1], clip1k(dy * cm));
      atomicAdd(&nxa[node * 4 + 2], clip1k(dz * cm));
    }
  }

  WAIT_DS();
  if (lane < 24) {
    const int node = lane / 3, d = lane - node * 3;
    const int gn = g * 8 + node;
    xB[gn * 3 + d] = clip1k(xA[gn * 3 + d]) + nxa[node * 4 + d] * 0.1f;
  }
#pragma unroll
  for (int node = 0; node < 8; ++node) {
    float4 s4 = efl[node * 32 + lane];
    *(float4*)(efSum + (size_t)(g * 8 + node) * 128 + lane * 4) = s4;
  }
}

// ---------------- node MLP kernel ----------------
// LDS: 0: n1t bf16[128][256]; 65536: n2t bf16[128][128]; 98304: b1; 98816: b2; 99328: stg[8][16][128]
__global__ void __launch_bounds__(256)
egnn_node_kernel(const float* __restrict__ hhA, const float* __restrict__ efSum,
                 float* __restrict__ hhB,
                 const float* __restrict__ nw1, const float* __restrict__ nb1g,
                 const float* __restrict__ nw2, const float* __restrict__ nb2g) {
  extern __shared__ char smem[];
  __bf16* w1t = (__bf16*)(smem);
  __bf16* w2t = (__bf16*)(smem + 65536);
  float* b1 = (float*)(smem + 98304);
  float* b2 = (float*)(smem + 98816);
  __bf16* stgA = (__bf16*)(smem + 99328);

  const int tid = threadIdx.x;
  for (int i = tid; i < 128 * 256; i += 256) {
    int n = i >> 8, k = i & 255;
    w1t[i] = (__bf16)nw1[k * 128 + n];
  }
  for (int i = tid; i < 128 * 128; i += 256) {
    int n = i >> 7, k = i & 127;
    w2t[i] = (__bf16)nw2[k * 128 + n];
  }
  if (tid < 128) { b1[tid] = nb1g[tid]; b2[tid] = nb2g[tid]; }
  __syncthreads();

  const int wv = tid >> 5;
  const int lane = tid & 31;
  const int m = lane & 15;
  const int half = lane >> 4;
  __bf16* stg = stgA + wv * (16 * 128);
  const int row0 = (blockIdx.x * 8 + wv) * 16;  // 16 nodes per wave

  v8f acc[8];
#pragma unroll
  for (int nt = 0; nt < 8; ++nt) {
    const float b0 = b1[nt * 16 + m];
#pragma unroll
    for (int v = 0; v < 8; ++v) acc[nt][v] = b0;
  }
  const int node = row0 + m;
#pragma unroll
  for (int s = 0; s < 8; ++s) {
    const int kb = s * 32 + half * 8;
    const int j0 = kb, j1 = kb + 16;
    const float* p0 = (j0 < 128) ? (hhA + (size_t)node * 128 + j0)
                                 : (efSum + (size_t)node * 128 + (j0 - 128));
    const float* p1 = (j1 < 128) ? (hhA + (size_t)node * 128 + j1)
                                 : (efSum + (size_t)node * 128 + (j1 - 128));
    v16bf a = gblFragF32(p0, p1);
    v16bf bfr[8];
#pragma unroll
    for (int nt = 0; nt < 8; ++nt) bfr[nt] = ldsFragRow(w1t + (nt * 16 + m) * 256, kb);
#pragma unroll
    for (int nt = 0; nt < 8; ++nt) acc[nt] = wmma_bf16(a, bfr[nt], acc[nt]);
  }
#pragma unroll
  for (int nt = 0; nt < 8; ++nt) {
    const int n = nt * 16 + m;
#pragma unroll
    for (int v = 0; v < 8; ++v) stg[(v + 8 * half) * 128 + n] = (__bf16)siluf(acc[nt][v]);
  }
  WAIT_DS();
#pragma unroll
  for (int nt = 0; nt < 8; ++nt) {
    const float b0 = b2[nt * 16 + m];
#pragma unroll
    for (int v = 0; v < 8; ++v) acc[nt][v] = b0;
  }
#pragma unroll
  for (int s = 0; s < 4; ++s) {
    const int kb = s * 32 + half * 8;
    v16bf a = ldsFragRow(stg + m * 128, kb);
    v16bf bfr[8];
#pragma unroll
    for (int nt = 0; nt < 8; ++nt) bfr[nt] = ldsFragRow(w2t + (nt * 16 + m) * 128, kb);
#pragma unroll
    for (int nt = 0; nt < 8; ++nt) acc[nt] = wmma_bf16(a, bfr[nt], acc[nt]);
  }
  // residual add + store
#pragma unroll
  for (int nt = 0; nt < 8; ++nt) {
    const int n = nt * 16 + m;
#pragma unroll
    for (int v = 0; v < 8; ++v) {
      const int nd = row0 + v + 8 * half;
      hhB[(size_t)nd * 128 + n] = acc[nt][v] + hhA[(size_t)nd * 128 + n];
    }
  }
}

// ---------------- tiny kernels ----------------

__global__ void egnn_prep_kernel(const float* __restrict__ h,
                                 const float* __restrict__ w, const float* __restrict__ b,
                                 float* __restrict__ x0, float* __restrict__ hh0) {
  const int n = blockIdx.x;
  const int j = threadIdx.x;  // 128
  const float* hr = h + (size_t)n * 11;
  float acc = b[j];
#pragma unroll
  for (int i = 0; i < 8; ++i) acc += hr[3 + i] * w[i * 128 + j];
  hh0[(size_t)n * 128 + j] = acc;
  if (j < 3) x0[n * 3 + j] = hr[j] * (1.f / 3330.f);
}

__global__ void egnn_out_kernel(const float* __restrict__ hh, const float* __restrict__ x,
                                const float* __restrict__ w, const float* __restrict__ b,
                                float* __restrict__ out) {
  const int t = blockIdx.x * 256 + threadIdx.x;
  if (t >= NNODES * 4) return;
  const int n = t >> 2, o = t & 3;
  float acc = b[o];
#pragma unroll 4
  for (int j = 0; j < 128; ++j) acc += hh[(size_t)n * 128 + j] * w[j * 4 + o];
#pragma unroll
  for (int d = 0; d < 3; ++d) acc += x[n * 3 + d] * w[(128 + d) * 4 + o];
  out[t] = acc;
}

// ---------------- launcher ----------------

extern "C" void kernel_launch(void* const* d_in, const int* in_sizes, int n_in,
                              void* d_out, int out_size, void* d_ws, size_t ws_size,
                              hipStream_t stream) {
  (void)in_sizes; (void)n_in; (void)out_size; (void)ws_size;
  const float* h        = (const float*)d_in[0];
  const int*   idx      = (const int*)d_in[1];
  const float* emb_in_w = (const float*)d_in[2];
  const float* emb_in_b = (const float*)d_in[3];
  const float* edge_w1  = (const float*)d_in[4];
  const float* edge_b1  = (const float*)d_in[5];
  const float* edge_w2  = (const float*)d_in[6];
  const float* edge_b2  = (const float*)d_in[7];
  const float* coord_w1 = (const float*)d_in[8];
  const float* coord_b1 = (const float*)d_in[9];
  const float* coord_w2 = (const float*)d_in[10];
  const float* node_w1  = (const float*)d_in[11];
  const float* node_b1  = (const float*)d_in[12];
  const float* node_w2  = (const float*)d_in[13];
  const float* node_b2  = (const float*)d_in[14];
  const float* emb_out_w = (const float*)d_in[15];
  const float* emb_out_b = (const float*)d_in[16];
  float* out = (float*)d_out;

  char* ws = (char*)d_ws;
  float* xb[2]  = {(float*)ws, (float*)(ws + 196608)};
  float* hhb[2] = {(float*)(ws + 393216), (float*)(ws + 393216 + 8388608)};
  float* efSum  = (float*)(ws + 393216 + 2 * 8388608);

  egnn_prep_kernel<<<NNODES, 128, 0, stream>>>(h, emb_in_w, emb_in_b, xb[0], hhb[0]);

  for (int l = 0; l < NLAYERS; ++l) {
    const int a = l & 1, c = (l + 1) & 1;
    egnn_edge_kernel<<<256, 256, EDGE_LDS, stream>>>(
        xb[a], xb[c], hhb[a], efSum, idx,
        edge_w1 + (size_t)l * 257 * 128, edge_b1 + l * 128,
        edge_w2 + (size_t)l * 128 * 128, edge_b2 + l * 128,
        coord_w1 + (size_t)l * 128 * 128, coord_b1 + l * 128,
        coord_w2 + l * 128);
    egnn_node_kernel<<<128, 256, NODE_LDS, stream>>>(
        hhb[a], efSum, hhb[c],
        node_w1 + (size_t)l * 256 * 128, node_b1 + l * 128,
        node_w2 + (size_t)l * 128 * 128, node_b2 + l * 128);
  }

  egnn_out_kernel<<<(NNODES * 4 + 255) / 256, 256, 0, stream>>>(
      hhb[NLAYERS & 1], xb[NLAYERS & 1], emb_out_w, emb_out_b, out);
}
